// FusionInterp3D_90323162235018
// MI455X (gfx1250) — matrix operand order repaired and checked by reading, not compiled, not run on hardware
//
#include <hip/hip_runtime.h>
#include <hip/hip_bf16.h>
#include <math.h>

// Problem constants (match reference)
#define B_   2
#define C3_  64
#define H_   256
#define W_   512
#define N_   (H_ * W_)      // 131072 points per batch
#define K_   4
#define SLOPE 0.1f

typedef __attribute__((ext_vector_type(16))) __bf16 v16bf;
typedef __attribute__((ext_vector_type(8)))  float  v8f;

static __device__ __forceinline__ unsigned short f2bf(float f) {
  unsigned int u = __float_as_uint(f);
  u += 0x7FFFu + ((u >> 16) & 1u);           // round-to-nearest-even
  return (unsigned short)(u >> 16);
}
static __device__ __forceinline__ float leaky(float x) {
  return x >= 0.0f ? x : SLOPE * x;
}

// ---------------------------------------------------------------------------
// Kernel 0: convert w3 (64x64 f32) to bf16 once. Amortized over the whole
// dispatch so the WMMA kernel never spends VALU on f32->bf16 rounding.
// ---------------------------------------------------------------------------
__global__ void cvt_w3_kernel(const float* __restrict__ w3,
                              unsigned short* __restrict__ w3_bf) {
  const int i = blockIdx.x * 256 + threadIdx.x;   // 4096 elements
  w3_bf[i] = f2bf(w3[i]);
}

// ---------------------------------------------------------------------------
// Kernel 1: pack. Transpose feat_3d [b,c,n] -> f3_pk [b,n,c] (point-major so
// each KNN gather reads 256 contiguous bytes instead of 64 strided lines),
// and uv [b,2,n] -> uv_pk [b,n,2]. LDS-tiled for coalescing both directions.
// ---------------------------------------------------------------------------
__global__ void pack_kernel(const float* __restrict__ uv,
                            const float* __restrict__ feat3d,
                            float* __restrict__ uv_pk,
                            float* __restrict__ f3_pk) {
  __shared__ float tile[64][65];             // +1 pad: conflict-free both ways
  const int t   = threadIdx.x;               // 256 threads
  const int blk = blockIdx.x;                // B * (N/64) blocks
  const int b   = blk / (N_ / 64);
  const int n0  = (blk % (N_ / 64)) * 64;

  // coalesced read along n
  const float* src = feat3d + (size_t)b * C3_ * N_ + n0;
  const int nin = t & 63, cq = t >> 6;
#pragma unroll
  for (int r = 0; r < 16; ++r) {
    const int c = r * 4 + cq;
    tile[c][nin] = src[(size_t)c * N_ + nin];
  }
  if (t < 128) {                             // uv transpose (tiny)
    const int ch = t >> 6, j = t & 63;
    uv_pk[((size_t)b * N_ + n0 + j) * 2 + ch] =
        uv[((size_t)b * 2 + ch) * N_ + n0 + j];
  }
  __syncthreads();

  // coalesced write along c
  const int co = t & 63, nq = t >> 6;
  float* dst = f3_pk + ((size_t)b * N_ + n0) * 64;
#pragma unroll
  for (int r = 0; r < 16; ++r) {
    const int n = r * 4 + nq;
    dst[(size_t)n * 64 + co] = tile[co][n];
  }
}

// ---------------------------------------------------------------------------
// Kernel 2: gather + tiny MLP + score-weighted neighbor sum.
// Thread = (point, channel): 256 threads = 4 points x 64 channels.
// Writes final as bf16, point-major, ready for WMMA B-fragment loads.
// ---------------------------------------------------------------------------
__global__ void gather_mlp_kernel(const float* __restrict__ uv_pk,
                                  const float* __restrict__ f3_pk,
                                  const int* __restrict__ knn,
                                  const float* __restrict__ w1,
                                  const float* __restrict__ b1,
                                  const float* __restrict__ w2,
                                  const float* __restrict__ b2,
                                  unsigned short* __restrict__ final_bf) {
  __shared__ float sW1[48];
  __shared__ float sB1[16];
  __shared__ float sW2[64 * 17];             // pad 16->17: no bank conflicts
  __shared__ float sB2[64];
  const int t = threadIdx.x;
  for (int i = t; i < 48; i += 256) sW1[i] = w1[i];
  for (int i = t; i < 16; i += 256) sB1[i] = b1[i];
  for (int i = t; i < 1024; i += 256) sW2[(i >> 4) * 17 + (i & 15)] = w2[i];
  for (int i = t; i < 64; i += 256) sB2[i] = b2[i];
  __syncthreads();

  const int c = t & 63;
  const int p = blockIdx.x * 4 + (t >> 6);   // global point in [0, B*N)
  const int b = p / N_;
  const int n = p - b * N_;
  const float gx = (float)(n % W_);
  const float gy = (float)(n / W_);

  const float* uvb = uv_pk + (size_t)b * N_ * 2;
  const float* f3b = f3_pk + (size_t)b * N_ * 64;
  const int* kp = knn + (size_t)p * K_;

  float acc = 0.0f;
#pragma unroll
  for (int k = 0; k < K_; ++k) {
    const int idx = kp[k];
    const float ox = uvb[(size_t)idx * 2 + 0] - gx;
    const float oy = uvb[(size_t)idx * 2 + 1] - gy;
    const float nr = sqrtf(ox * ox + oy * oy);
    float s = sB2[c];
#pragma unroll
    for (int j = 0; j < 16; ++j) {           // hid_j recomputed per thread: cheap VALU
      float h = sW1[j * 3 + 0] * ox + sW1[j * 3 + 1] * oy +
                sW1[j * 3 + 2] * nr + sB1[j];
      h = leaky(h);
      s += sW2[c * 17 + j] * h;
    }
    const float score = 1.0f / (1.0f + __expf(-s));
    acc += score * f3b[(size_t)idx * 64 + c];   // 256B-contiguous gather
  }
  final_bf[(size_t)p * 64 + c] = f2bf(acc);
}

// ---------------------------------------------------------------------------
// Kernel 3: out = leaky(w3 @ final + b3) via v_wmma_f32_16x16x32_bf16.
// One wave32 owns 64 output channels and loops (NOT unrolled) over TPW=8
// point-tiles of 16 points. __launch_bounds__(256,1) lifts the VGPR cap so
// the 8 persistent A fragments + bias + acc (~190 VGPRs) fit with NO spills
// (2 waves/SIMD; final_bf fits in 192MB L2 so latency is short anyway).
// Bias enters as the C operand of the first WMMA of each chain: no copies.
// ---------------------------------------------------------------------------
#define TPW 8   // point-tiles per wave (8*16 = 128 points, batch-aligned)

__global__ void __launch_bounds__(256, 1)
wmma_out_kernel(const unsigned short* __restrict__ final_bf,
                const unsigned short* __restrict__ w3_bf,
                const float* __restrict__ b3,
                float* __restrict__ out) {
  __shared__ __align__(16) unsigned short sW3[64 * 64];  // 8 KB bf16
  __shared__ float sB3[64];
  const int t = threadIdx.x;                 // 256 threads = 8 waves
  for (int i = t; i < 2048; i += 256)        // coalesced dword copy
    ((unsigned int*)sW3)[i] = ((const unsigned int*)w3_bf)[i];
  if (t < 64) sB3[t] = b3[t];
  __syncthreads();

  const int lane = t & 31;
  const int wid  = t >> 5;
  const int g    = lane >> 4;                // lane group (0/1)
  const int m    = lane & 15;

  // A fragments: 16-bit A 16x32, k in contiguous order (not the ISA table order).
  // lane (g,m): VGPR0-7 pack K = g*16+0..15 (contiguous, matches B).
  v16bf A[4][2];
#pragma unroll
  for (int ct = 0; ct < 4; ++ct) {
    const int row = ct * 16 + m;
#pragma unroll
    for (int kk = 0; kk < 2; ++kk) {
      const unsigned short* r0 = &sW3[row * 64 + kk * 32 + g * 16];
      union { uint4 q[2]; v16bf v; } fa;
      fa.q[0] = *(const uint4*)(r0);         // 16B-aligned ds_load_b128
      fa.q[1] = *(const uint4*)(r0 + 8);
      A[ct][kk] = fa.v;
    }
  }

  // Bias pre-broadcast: row co = ct*16 + r + 8*g gets bias b3[co] in every
  // column, so it can serve directly as the WMMA C-matrix initializer.
  v8f bias_v[4];
#pragma unroll
  for (int ct = 0; ct < 4; ++ct)
#pragma unroll
    for (int r = 0; r < 8; ++r) bias_v[ct][r] = sB3[ct * 16 + r + 8 * g];

  // This wave's 128 consecutive points: all in one batch (N % 128 == 0).
  const int p0base = (blockIdx.x * 8 + wid) * (TPW * 16);
  const int b  = p0base / N_;
  const int n0 = p0base - b * N_;

  // B-fragment pointer (lane-dependent), advanced by 16 points per tile.
  const unsigned short* bp = final_bf + ((size_t)(p0base + m) * 64 + g * 16);

  // Per-ct output base pointers; row offset r*N_ is an immediate (<8MB).
  float* op0 = out + ((size_t)(b * 64 + 0 * 16 + 8 * g)) * N_ + n0 + m;
  float* op1 = out + ((size_t)(b * 64 + 1 * 16 + 8 * g)) * N_ + n0 + m;
  float* op2 = out + ((size_t)(b * 64 + 2 * 16 + 8 * g)) * N_ + n0 + m;
  float* op3 = out + ((size_t)(b * 64 + 3 * 16 + 8 * g)) * N_ + n0 + m;

#pragma unroll 1
  for (int tt = 0; tt < TPW; ++tt) {
    // B fragments: dense 16-bit B 32x16: lanes 0-15 K=0..15, lanes 16-31
    // K=16..31, N = lane%16; K contiguous per lane -> 32B contiguous load.
    union { uint4 q[2]; v16bf v; } fb0, fb1;
    fb0.q[0] = *(const uint4*)(bp);
    fb0.q[1] = *(const uint4*)(bp + 8);
    fb1.q[0] = *(const uint4*)(bp + 32);     // K-slice 32..63
    fb1.q[1] = *(const uint4*)(bp + 40);

    // acc chains start from bias_v as the C operand (no register copies;
    // D is allocated separately, bias_v survives for the next tile).
    v8f acc[4];
#pragma unroll
    for (int ct = 0; ct < 4; ++ct) {
      acc[ct] = __builtin_amdgcn_wmma_f32_16x16x32_bf16(
          false, A[ct][0], false, fb0.v, (short)0, bias_v[ct], false, false);
      acc[ct] = __builtin_amdgcn_wmma_f32_16x16x32_bf16(
          false, A[ct][1], false, fb1.v, (short)0, acc[ct], false, false);
    }

    // Epilogue: D layout m = vgpr + 8*(lane/16), n = lane%16.
    // Immediate-offset stores: offset = r*N_ floats (constant per unrolled r).
#pragma unroll
    for (int r = 0; r < 8; ++r) op0[(size_t)r * N_] = leaky(acc[0][r]);
#pragma unroll
    for (int r = 0; r < 8; ++r) op1[(size_t)r * N_] = leaky(acc[1][r]);
#pragma unroll
    for (int r = 0; r < 8; ++r) op2[(size_t)r * N_] = leaky(acc[2][r]);
#pragma unroll
    for (int r = 0; r < 8; ++r) op3[(size_t)r * N_] = leaky(acc[3][r]);

    bp  += 16 * 64;                          // next 16 points
    op0 += 16; op1 += 16; op2 += 16; op3 += 16;
  }
}

// ---------------------------------------------------------------------------
extern "C" void kernel_launch(void* const* d_in, const int* in_sizes, int n_in,
                              void* d_out, int out_size, void* d_ws, size_t ws_size,
                              hipStream_t stream) {
  const float* uv     = (const float*)d_in[0];
  // d_in[1] = feat_2d: unused by the reference math (shape only)
  const float* feat3d = (const float*)d_in[2];
  const int*   knn    = (const int*)d_in[3];
  const float* w1     = (const float*)d_in[4];
  const float* b1     = (const float*)d_in[5];
  const float* w2     = (const float*)d_in[6];
  const float* b2     = (const float*)d_in[7];
  const float* w3     = (const float*)d_in[8];
  const float* b3     = (const float*)d_in[9];
  float* out = (float*)d_out;

  // Workspace: f3_pk (B*N*64 f32) | uv_pk (B*N*2 f32) | final (B*N*64 bf16)
  //            | w3_bf (64*64 bf16)
  char* ws = (char*)d_ws;
  float* f3_pk = (float*)ws;
  float* uv_pk = (float*)(ws + (size_t)B_ * N_ * 64 * 4);
  unsigned short* final_bf =
      (unsigned short*)(ws + (size_t)B_ * N_ * 64 * 4 + (size_t)B_ * N_ * 2 * 4);
  unsigned short* w3_bf = final_bf + (size_t)B_ * N_ * 64;

  cvt_w3_kernel<<<16, 256, 0, stream>>>(w3, w3_bf);
  pack_kernel<<<B_ * (N_ / 64), 256, 0, stream>>>(uv, feat3d, uv_pk, f3_pk);
  gather_mlp_kernel<<<(B_ * N_) / 4, 256, 0, stream>>>(
      uv_pk, f3_pk, knn, w1, b1, w2, b2, final_bf);
  // tiles = B*N/16 = 16384; 8 waves x TPW tiles per block -> 256 blocks
  wmma_out_kernel<<<(B_ * N_ / 16) / (8 * TPW), 256, 0, stream>>>(
      final_bf, w3_bf, b3, out);
}
